// GraphWaveNet_48722109005840
// MI455X (gfx1250) — matrix operand order repaired and checked by reading, not compiled, not run on hardware
//
#include <hip/hip_runtime.h>
#include <cmath>

namespace {
constexpr int kB = 16, kL = 13, kN = 1024, kR = 32, kD = 32, kS = 256,
              kE = 512, kO = 12, kEmb = 10, kLayers = 8;
constexpr int kCol = kB * kN;              // 16384 head columns (b,n)
constexpr float kBnEps = 1e-5f;
}

typedef _Float16 f16;
typedef __attribute__((ext_vector_type(16))) _Float16 v16h;
typedef __attribute__((ext_vector_type(8)))  float    v8f;
typedef __attribute__((ext_vector_type(4)))  int      v4i;

union V16H { v16h h; v4i i[2]; };

__device__ inline v16h ld_frag(const f16* p0, const f16* p1) {
  V16H u;
  u.i[0] = *(const v4i*)p0;   // 8 f16, 16B aligned
  u.i[1] = *(const v4i*)p1;   // 8 f16, 16B aligned
  return u.h;
}

// ---------------------------------------------------------------------------
// adp = softmax(relu(E E^T), axis=1); emit split-f16 TRANSPOSED copies:
//   adpT_hi[w*1024+v] + adpT_lo[w*1024+v]  ~=  adp[v][w]  (to ~2^-24)
// ---------------------------------------------------------------------------
__global__ __launch_bounds__(256)
void gwn_adp(const float* __restrict__ emb, f16* __restrict__ adpT_hi,
             f16* __restrict__ adpT_lo) {
  __shared__ float red[256];
  __shared__ float ev[kEmb];
  const int v = blockIdx.x, t = threadIdx.x;
  if (t < kEmb) ev[t] = emb[v * kEmb + t];
  __syncthreads();
  float s[4];
  float lmax = -3.4e38f;
#pragma unroll
  for (int j = 0; j < 4; ++j) {
    const int w = t + j * 256;
    float acc = 0.f;
#pragma unroll
    for (int k = 0; k < kEmb; ++k) acc += ev[k] * emb[w * kEmb + k];
    acc = fmaxf(acc, 0.f);
    s[j] = acc;
    lmax = fmaxf(lmax, acc);
  }
  red[t] = lmax; __syncthreads();
  for (int o = 128; o > 0; o >>= 1) {
    if (t < o) red[t] = fmaxf(red[t], red[t + o]);
    __syncthreads();
  }
  const float m = red[0];
  __syncthreads();
  float lsum = 0.f;
#pragma unroll
  for (int j = 0; j < 4; ++j) { s[j] = expf(s[j] - m); lsum += s[j]; }
  red[t] = lsum; __syncthreads();
  for (int o = 128; o > 0; o >>= 1) {
    if (t < o) red[t] += red[t + o];
    __syncthreads();
  }
  const float inv = 1.f / red[0];
#pragma unroll
  for (int j = 0; j < 4; ++j) {
    const int w = t + j * 256;
    const float p = s[j] * inv;
    const f16 hi = (f16)p;
    adpT_hi[(size_t)w * kN + v] = hi;
    adpT_lo[(size_t)w * kN + v] = (f16)(p - (float)hi);
  }
}

// ---------------------------------------------------------------------------
// start conv: h[b][l][c][n] = sw[c][0]*x[b][l][n][0] + sw[c][1]*x[b][l][n][1] + sb[c]
// ---------------------------------------------------------------------------
__global__ __launch_bounds__(256)
void gwn_start(const float* __restrict__ x, const float* __restrict__ sw,
               const float* __restrict__ sb, float* __restrict__ h) {
  const int idx = blockIdx.x * 256 + threadIdx.x;   // ((b*L+l)*32+c)*1024+n
  const int n = idx & (kN - 1);
  const int c = (idx >> 10) & 31;
  const int bl = idx >> 15;                         // b*L + l
  const float x0 = x[(bl * kN + n) * 2 + 0];
  const float x1 = x[(bl * kN + n) * 2 + 1];
  h[idx] = sw[c * 2 + 0] * x0 + sw[c * 2 + 1] * x1 + sb[c];
}

// ---------------------------------------------------------------------------
// dilated 2-tap conv + gate, 4 output channels per thread (4x load reuse)
// gate via hardware v_exp_f32 (clamped; |err| ~ 1e-6)
// emits split-f16 g:  g_hi + g_lo ~= g   (layout [B][Lg][32][N])
// ---------------------------------------------------------------------------
__global__ __launch_bounds__(256)
void gwn_tcn_gate(const float* __restrict__ h, const float* __restrict__ tw,
                  const float* __restrict__ tb, f16* __restrict__ ghi,
                  f16* __restrict__ glo, int Lc, int Lg, int d) {
  const int idx = blockIdx.x * 256 + threadIdx.x;   // ((b*Lg+l)*8+c4)*1024+n
  const int n = idx & (kN - 1);
  const int c4 = (idx >> 10) & 7;
  const int bl = idx >> 13;
  const int b = bl / Lg;
  const int l = bl - b * Lg;
  const float* h0 = h + (((size_t)(b * Lc + l) * kR) << 10) + n;       // tap l
  const float* h1 = h + (((size_t)(b * Lc + l + d) * kR) << 10) + n;   // tap l+d
  float acc[4] = {tb[c4], tb[c4 + 8], tb[c4 + 16], tb[c4 + 24]};
#pragma unroll
  for (int c = 0; c < kR; ++c) {
    const float v0 = h0[c << 10];
    const float v1 = h1[c << 10];
#pragma unroll
    for (int j = 0; j < 4; ++j) {
      const int dc = c4 + 8 * j;
      acc[j] += tw[(dc * kR + c) * 2 + 0] * v0 + tw[(dc * kR + c) * 2 + 1] * v1;
    }
  }
#pragma unroll
  for (int j = 0; j < 4; ++j) {
    const int dc = c4 + 8 * j;
    const float xc = fminf(fmaxf(acc[j], -15.f), 15.f);
    const float e2 = __expf(2.f * xc);
    const float gate = ((e2 - 1.f) / (e2 + 1.f)) * (1.f / (1.f + __expf(-xc)));
    const f16 hi = (f16)gate;
    const size_t oi = (((size_t)bl * kR + dc) << 10) + n;
    ghi[oi] = hi;
    glo[oi] = (f16)(gate - (float)hi);
  }
}

// ---------------------------------------------------------------------------
// skip contribution from last time index of g (g = hi + lo), 4 outputs/thread
//   skipacc[b][o][n] (init?=:+=) sum_c skip_w[o][c]*g[b][Lg-1][c][n] + skip_b[o]
// ---------------------------------------------------------------------------
__global__ __launch_bounds__(256)
void gwn_skip(const f16* __restrict__ ghi, const f16* __restrict__ glo,
              const float* __restrict__ sw, const float* __restrict__ sb,
              float* __restrict__ skipacc, int Lg, int init) {
  const int idx = blockIdx.x * 256 + threadIdx.x;   // ((b*64)+o4)*1024+n
  const int n = idx & (kN - 1);
  const int o4 = (idx >> 10) & 63;
  const int b = idx >> 16;
  const size_t gbase = (((size_t)(b * Lg + (Lg - 1)) * kD) << 10) + n;
  float acc[4] = {sb[o4], sb[o4 + 64], sb[o4 + 128], sb[o4 + 192]};
#pragma unroll
  for (int c = 0; c < kD; ++c) {
    const size_t gi = gbase + ((size_t)c << 10);
    const float gv = (float)ghi[gi] + (float)glo[gi];
#pragma unroll
    for (int j = 0; j < 4; ++j) acc[j] += sw[(o4 + 64 * j) * kD + c] * gv;
  }
#pragma unroll
  for (int j = 0; j < 4; ++j) {
    const size_t oi = (((size_t)b * kS + o4 + 64 * j) << 10) + n;
    skipacc[oi] = init ? acc[j] : (skipacc[oi] + acc[j]);
  }
}

// ---------------------------------------------------------------------------
// graph diffusion GEMM, split-f16 WMMA (3x v_wmma_f32_16x16x32_f16 per tile
// per K-step; f32 accumulate; ~fp32 accuracy):
//   gd[m][w] = sum_v g[m][v] * adp[v][w],  m=(b,l,c), K = N = 1024
// Wave tile 32x32; block (8 waves) 64x128; grid = (M/64, 1024/128)
// ---------------------------------------------------------------------------
__global__ __launch_bounds__(256)
void gwn_diffusion(const f16* __restrict__ ghi, const f16* __restrict__ glo,
                   const f16* __restrict__ bhiT, const f16* __restrict__ bloT,
                   float* __restrict__ gd) {
  const int lane = threadIdx.x & 31;
  const int wave = threadIdx.x >> 5;
  const int wm = wave >> 2;                 // 0..1
  const int wn = wave & 3;                  // 0..3
  const int m0 = blockIdx.x * 64 + wm * 32;
  const int n0 = blockIdx.y * 128 + wn * 32;
  const int half = lane >> 4;
  const int l16 = lane & 15;

  v8f acc[2][2] = {};
  for (int k0 = 0; k0 < kN; k0 += 32) {
    v16h ahi[2], alo[2];
#pragma unroll
    for (int mt = 0; mt < 2; ++mt) {
      const size_t rowoff = (size_t)(m0 + mt * 16 + l16) * kN + k0 + half * 8;
      ahi[mt] = ld_frag(ghi + rowoff, ghi + rowoff + 16);
      alo[mt] = ld_frag(glo + rowoff, glo + rowoff + 16);
    }
    v16h bhi[2], blo[2];
#pragma unroll
    for (int nt = 0; nt < 2; ++nt) {
      const size_t coloff = (size_t)(n0 + nt * 16 + l16) * kN + k0 + half * 8;
      bhi[nt] = ld_frag(bhiT + coloff, bhiT + coloff + 16);
      blo[nt] = ld_frag(bloT + coloff, bloT + coloff + 16);
    }
#pragma unroll
    for (int mt = 0; mt < 2; ++mt)
#pragma unroll
      for (int nt = 0; nt < 2; ++nt) {
        acc[mt][nt] = __builtin_amdgcn_wmma_f32_16x16x32_f16(
            false, alo[mt], false, bhi[nt], (short)0, acc[mt][nt], false, false);
        acc[mt][nt] = __builtin_amdgcn_wmma_f32_16x16x32_f16(
            false, ahi[mt], false, blo[nt], (short)0, acc[mt][nt], false, false);
        acc[mt][nt] = __builtin_amdgcn_wmma_f32_16x16x32_f16(
            false, ahi[mt], false, bhi[nt], (short)0, acc[mt][nt], false, false);
      }
  }
#pragma unroll
  for (int mt = 0; mt < 2; ++mt)
#pragma unroll
    for (int nt = 0; nt < 2; ++nt)
#pragma unroll
      for (int r = 0; r < 8; ++r) {
        const int row = m0 + mt * 16 + r + half * 8;
        const int col = n0 + nt * 16 + l16;
        gd[(size_t)row * kN + col] = acc[mt][nt][r];
      }
}

// ---------------------------------------------------------------------------
// gcn 1x1 + relu + residual(tail) + BatchNorm(eval), 4 output channels/thread
// ---------------------------------------------------------------------------
__global__ __launch_bounds__(256)
void gwn_gcn_res_bn(const float* __restrict__ gd, const float* __restrict__ hin,
                    const float* __restrict__ gw, const float* __restrict__ gb,
                    const float* __restrict__ gamma, const float* __restrict__ beta,
                    const float* __restrict__ mean, const float* __restrict__ var,
                    float* __restrict__ hout, int Lc, int Lg, int d) {
  const int idx = blockIdx.x * 256 + threadIdx.x;   // ((b*Lg+l)*8+c4)*1024+n
  const int n = idx & (kN - 1);
  const int c4 = (idx >> 10) & 7;
  const int bl = idx >> 13;
  const int b = bl / Lg;
  const int l = bl - b * Lg;
  const float* gp = gd + (((size_t)bl * kD) << 10) + n;
  float acc[4] = {gb[c4], gb[c4 + 8], gb[c4 + 16], gb[c4 + 24]};
#pragma unroll
  for (int cc = 0; cc < kD; ++cc) {
    const float gv = gp[cc << 10];
#pragma unroll
    for (int j = 0; j < 4; ++j) acc[j] += gw[(c4 + 8 * j) * kD + cc] * gv;
  }
#pragma unroll
  for (int j = 0; j < 4; ++j) {
    const int c = c4 + 8 * j;
    const float r = fmaxf(acc[j], 0.f);
    const float res =
        hin[(((size_t)(b * Lc + l + d) * kR + c) << 10) + n];
    const float val = r + res;
    const float inv = gamma[c] * rsqrtf(var[c] + kBnEps);
    hout[(((size_t)bl * kR + c) << 10) + n] = (val - mean[c]) * inv + beta[c];
  }
}

// ---------------------------------------------------------------------------
// head prep 1: split end1_w (512x256, K contiguous) into f16 hi/lo
// ---------------------------------------------------------------------------
__global__ __launch_bounds__(256)
void gwn_split_w1(const float* __restrict__ w1, f16* __restrict__ w1hi,
                  f16* __restrict__ w1lo) {
  const int idx = blockIdx.x * 256 + threadIdx.x;   // 512*256
  const float v = w1[idx];
  const f16 hi = (f16)v;
  w1hi[idx] = hi;
  w1lo[idx] = (f16)(v - (float)hi);
}

// ---------------------------------------------------------------------------
// head prep 2: skT[col][c] = split-f16 of relu(skipacc[b][c][n]), col=b*1024+n
// ---------------------------------------------------------------------------
__global__ __launch_bounds__(256)
void gwn_skT(const float* __restrict__ skipacc, f16* __restrict__ skThi,
             f16* __restrict__ skTlo) {
  const int idx = blockIdx.x * 256 + threadIdx.x;   // ((b*256)+o)*1024+n
  const int n = idx & (kN - 1);
  const int o = (idx >> 10) & 255;
  const int b = idx >> 18;
  const float v = fmaxf(skipacc[idx], 0.f);
  const f16 hi = (f16)v;
  const size_t oi = ((size_t)(b * kN + n)) * kS + o;
  skThi[oi] = hi;
  skTlo[oi] = (f16)(v - (float)hi);
}

// ---------------------------------------------------------------------------
// head end1 GEMM, split-f16 WMMA:  e1[e][col] = relu(W1[e][:]·skrelu[:][col]+b1)
//   M = 512, K = 256, Ncol = 16384.  Wave tile 32x32; grid (8, 128)
// ---------------------------------------------------------------------------
__global__ __launch_bounds__(256)
void gwn_end1(const f16* __restrict__ w1hi, const f16* __restrict__ w1lo,
              const f16* __restrict__ skThi, const f16* __restrict__ skTlo,
              const float* __restrict__ b1, float* __restrict__ e1) {
  const int lane = threadIdx.x & 31;
  const int wave = threadIdx.x >> 5;
  const int wm = wave >> 2;
  const int wn = wave & 3;
  const int m0 = blockIdx.x * 64 + wm * 32;
  const int n0 = blockIdx.y * 128 + wn * 32;
  const int half = lane >> 4;
  const int l16 = lane & 15;

  v8f acc[2][2] = {};
#pragma unroll
  for (int k0 = 0; k0 < kS; k0 += 32) {
    v16h ahi[2], alo[2];
#pragma unroll
    for (int mt = 0; mt < 2; ++mt) {
      const size_t rowoff = (size_t)(m0 + mt * 16 + l16) * kS + k0 + half * 8;
      ahi[mt] = ld_frag(w1hi + rowoff, w1hi + rowoff + 16);
      alo[mt] = ld_frag(w1lo + rowoff, w1lo + rowoff + 16);
    }
    v16h bhi[2], blo[2];
#pragma unroll
    for (int nt = 0; nt < 2; ++nt) {
      const size_t coloff = (size_t)(n0 + nt * 16 + l16) * kS + k0 + half * 8;
      bhi[nt] = ld_frag(skThi + coloff, skThi + coloff + 16);
      blo[nt] = ld_frag(skTlo + coloff, skTlo + coloff + 16);
    }
#pragma unroll
    for (int mt = 0; mt < 2; ++mt)
#pragma unroll
      for (int nt = 0; nt < 2; ++nt) {
        acc[mt][nt] = __builtin_amdgcn_wmma_f32_16x16x32_f16(
            false, alo[mt], false, bhi[nt], (short)0, acc[mt][nt], false, false);
        acc[mt][nt] = __builtin_amdgcn_wmma_f32_16x16x32_f16(
            false, ahi[mt], false, blo[nt], (short)0, acc[mt][nt], false, false);
        acc[mt][nt] = __builtin_amdgcn_wmma_f32_16x16x32_f16(
            false, ahi[mt], false, bhi[nt], (short)0, acc[mt][nt], false, false);
      }
  }
#pragma unroll
  for (int mt = 0; mt < 2; ++mt)
#pragma unroll
    for (int nt = 0; nt < 2; ++nt)
#pragma unroll
      for (int r = 0; r < 8; ++r) {
        const int row = m0 + mt * 16 + r + half * 8;
        const int col = n0 + nt * 16 + l16;
        e1[(size_t)row * kCol + col] = fmaxf(acc[mt][nt][r] + b1[row], 0.f);
      }
}

// ---------------------------------------------------------------------------
// head end2: out[b][n][o] = sum_e w2[o][e]*e1[e][col] + b2[o]   (12x512, VALU)
// one thread per column; weights are wave-uniform (scalarizable)
// ---------------------------------------------------------------------------
__global__ __launch_bounds__(256)
void gwn_end2(const float* __restrict__ e1, const float* __restrict__ w2,
              const float* __restrict__ b2, float* __restrict__ out) {
  const int col = blockIdx.x * 256 + threadIdx.x;   // 0..16383
  float acc[kO];
#pragma unroll
  for (int o = 0; o < kO; ++o) acc[o] = b2[o];
  for (int e = 0; e < kE; ++e) {
    const float x = e1[(size_t)e * kCol + col];
#pragma unroll
    for (int o = 0; o < kO; ++o) acc[o] += w2[o * kE + e] * x;
  }
#pragma unroll
  for (int o = 0; o < kO; ++o) out[(size_t)col * kO + o] = acc[o];
}

// ---------------------------------------------------------------------------
extern "C" void kernel_launch(void* const* d_in, const int* in_sizes, int n_in,
                              void* d_out, int out_size, void* d_ws, size_t ws_size,
                              hipStream_t stream) {
  (void)in_sizes; (void)n_in; (void)out_size; (void)ws_size;
  const float* x        = (const float*)d_in[0];
  const float* node_emb = (const float*)d_in[1];
  const float* start_w  = (const float*)d_in[2];
  const float* start_b  = (const float*)d_in[3];
  const float* tcn_w    = (const float*)d_in[4];
  const float* tcn_b    = (const float*)d_in[5];
  const float* skip_w   = (const float*)d_in[6];
  const float* skip_b   = (const float*)d_in[7];
  const float* gcn_w    = (const float*)d_in[8];
  const float* gcn_b    = (const float*)d_in[9];
  const float* bn_gamma = (const float*)d_in[10];
  const float* bn_beta  = (const float*)d_in[11];
  const float* bn_mean  = (const float*)d_in[12];
  const float* bn_var   = (const float*)d_in[13];
  const float* end1_w   = (const float*)d_in[14];
  const float* end1_b   = (const float*)d_in[15];
  const float* end2_w   = (const float*)d_in[16];
  const float* end2_b   = (const float*)d_in[17];
  float* out = (float*)d_out;

  // workspace carve-up: 120 MiB total
  f16* adpT_hi = (f16*)d_ws;                          // 1024*1024 halfs
  f16* adpT_lo = adpT_hi + (size_t)kN * kN;
  f16* ghi     = adpT_lo + (size_t)kN * kN;           // 16*12*32*1024 halfs
  f16* glo     = ghi + (size_t)kB * 12 * kD * kN;
  float* h_a     = (float*)(glo + (size_t)kB * 12 * kD * kN);
  float* h_b     = h_a + (size_t)kB * kL * kR * kN;   // 16*13*32*1024 floats
  float* gdbuf   = h_b + (size_t)kB * kL * kR * kN;   // 16*12*32*1024 floats
  float* skipacc = gdbuf + (size_t)kB * 12 * kD * kN; // 16*256*1024 floats

  // head-time aliases (g/h/gd regions are dead once the layer loop finishes)
  f16* w1hi  = ghi;                                   // 512*256 halfs
  f16* w1lo  = w1hi + (size_t)kE * kS;
  f16* skThi = (f16*)h_a;                             // 16384*256 halfs
  f16* skTlo = skThi + (size_t)kCol * kS;
  float* e1  = h_b;                                   // 512*16384 floats
                                                      // (spans h_b + gdbuf)

  gwn_adp<<<kN, 256, 0, stream>>>(node_emb, adpT_hi, adpT_lo);

  const int startTotal = kB * kL * kR * kN;
  gwn_start<<<startTotal / 256, 256, 0, stream>>>(x, start_w, start_b, h_a);

  const int dil[kLayers] = {1, 2, 1, 2, 1, 2, 1, 2};
  float* hcur = h_a;
  float* hnext = h_b;
  int Lc = kL;
  for (int i = 0; i < kLayers; ++i) {
    const int d = dil[i];
    const int Lg = Lc - d;

    gwn_tcn_gate<<<(kB * Lg * 8 * kN) / 256, 256, 0, stream>>>(
        hcur, tcn_w + (size_t)i * kD * kR * 2, tcn_b + i * kD,
        ghi, glo, Lc, Lg, d);

    gwn_skip<<<(kB * 64 * kN) / 256, 256, 0, stream>>>(
        ghi, glo, skip_w + (size_t)i * kS * kD, skip_b + i * kS, skipacc, Lg,
        (i == 0) ? 1 : 0);

    const int M = kB * Lg * kD;             // 512*Lg, divisible by 64
    dim3 grid5(M / 64, kN / 128);
    gwn_diffusion<<<grid5, 256, 0, stream>>>(ghi, glo, adpT_hi, adpT_lo, gdbuf);

    gwn_gcn_res_bn<<<(kB * Lg * 8 * kN) / 256, 256, 0, stream>>>(
        gdbuf, hcur, gcn_w + (size_t)i * kR * kD, gcn_b + i * kR,
        bn_gamma + i * kR, bn_beta + i * kR, bn_mean + i * kR, bn_var + i * kR,
        hnext, Lc, Lg, d);

    float* tmp = hcur; hcur = hnext; hnext = tmp;
    Lc = Lg;
  }

  // ---- head ----
  gwn_split_w1<<<(kE * kS) / 256, 256, 0, stream>>>(end1_w, w1hi, w1lo);
  gwn_skT<<<(kB * kS * kN) / 256, 256, 0, stream>>>(skipacc, skThi, skTlo);
  {
    dim3 gridE(kE / 64, kCol / 128);
    gwn_end1<<<gridE, 256, 0, stream>>>(w1hi, w1lo, skThi, skTlo, end1_b, e1);
  }
  gwn_end2<<<kCol / 256, 256, 0, stream>>>(e1, end2_w, end2_b, out);
}
